// MultiheadAttention_62448824484374
// MI455X (gfx1250) — compile-verified, matrix operand order checked by reading, not run on hardware
//
#include <hip/hip_runtime.h>
#include <hip/hip_bf16.h>
#include <math.h>
#include <stdint.h>

#define BB 4
#define TT 2048
#define DD 1024
#define HH 16
#define HD 64
#define BT (BB*TT)

typedef __bf16 bf16;
typedef __attribute__((ext_vector_type(16))) __bf16 v16bf;
typedef __attribute__((ext_vector_type(8)))  float   v8f;
typedef __attribute__((ext_vector_type(16))) float   v16f;
typedef __attribute__((ext_vector_type(4)))  unsigned int v4u;
typedef __attribute__((ext_vector_type(8)))  int     v8i_t;
typedef __attribute__((ext_vector_type(4)))  int     v4i_t;

static __device__ __forceinline__ v8f wmma_bf16(v16bf a, v16bf b, v8f c) {
  // 8 args: (neg_a, A, neg_b, B, c_mod, C, reuse_a, reuse_b)
  return __builtin_amdgcn_wmma_f32_16x16x32_bf16(false, a, false, b, (short)0, c,
                                                 false, false);
}

// Tensor Data Mover: 2D tile of bf16 (data_size=2B) global -> LDS.
// D# packing per CDNA5 ISA 8.3/8.4 (group0: count/lds/global/type=2;
// group1: data_size, tensor_dim0/1, tile_dim0/1, dim0 stride).
// This toolchain exposes the 6-arg builtin: (g0, g1, g2, g3, g_extra, cpol).
static __device__ __forceinline__ void tdm_load_2d_bf16(
    unsigned lds_byte_off, const bf16* gptr, unsigned tensor_w,
    unsigned tensor_h, unsigned tile_w, unsigned tile_h,
    unsigned stride_elems) {
  unsigned long long ga = (unsigned long long)(uintptr_t)gptr;
  v4u g0;
  g0.x = 1u;                                    // count=1 (valid), load
  g0.y = lds_byte_off;                          // lds_addr
  g0.z = (unsigned)(ga & 0xFFFFFFFFu);          // global_addr[31:0]
  g0.w = (unsigned)((ga >> 32) & 0x01FFFFFFu)   // global_addr[56:32]
         | (2u << 30);                          // type=2 ("image")
  v8i_t g1;
  g1[0] = (int)(1u << 16);                      // data_size=1 -> 2 bytes
  g1[1] = (int)((tensor_w & 0xFFFFu) << 16);    // tensor_dim0[15:0]
  g1[2] = (int)((tensor_w >> 16) | ((tensor_h & 0xFFFFu) << 16));
  g1[3] = (int)((tensor_h >> 16) | (tile_w << 16));   // tile_dim0
  g1[4] = (int)(tile_h & 0xFFFFu);              // tile_dim1 (tile_dim2=0)
  g1[5] = (int)stride_elems;                    // tensor_dim0_stride[31:0]
  g1[6] = 0;
  g1[7] = 0;
  v4i_t z4 = {0, 0, 0, 0};
  v8i_t z8 = {0, 0, 0, 0, 0, 0, 0, 0};
  __builtin_amdgcn_tensor_load_to_lds(g0, g1, z4, z4, z8, 0);
}

// ---------------------------------------------------------------- conversions
__global__ void cvt_bf16_kernel(const float* __restrict__ src,
                                bf16* __restrict__ dst, int n) {
  int i = blockIdx.x * blockDim.x + threadIdx.x;
  int stride = gridDim.x * blockDim.x;
  for (; i < n; i += stride) dst[i] = (bf16)src[i];
}

__global__ void zero_kernel(float* __restrict__ p, int n) {
  int i = blockIdx.x * blockDim.x + threadIdx.x;
  int stride = gridDim.x * blockDim.x;
  for (; i < n; i += stride) p[i] = 0.0f;
}

// ---------------------------------------------------------------- GEMM
// C[m,n] = (sum_k A[m,k]*W[n,k] + bias[n]) * scale
// TDM double-buffered staging: wave 0 issues tensor_load_to_lds for tile k+1
// while all waves compute tile k from LDS.
// mode 0: bf16 out [m][n];  mode 2: bf16 out vt[b,h,d,t];  mode 3: fp32 out.
__global__ __launch_bounds__(256) void gemm_proj_kernel(
    const bf16* __restrict__ A, const bf16* __restrict__ W,
    const float* __restrict__ bias,
    bf16* __restrict__ out_bf, float* __restrict__ out_f,
    float scale, int mode) {
  __shared__ bf16 At[2][128 * 32];
  __shared__ bf16 Bt[2][128 * 32];

  const int tid = threadIdx.x;
  const int lane = tid & 31;
  const int wave = tid >> 5;
  const int wm = wave & 3;   // wave row: m offset wm*32
  const int wn = wave >> 2;  // wave col: n offset wn*64
  const int m0 = blockIdx.y * 128;
  const int n0 = blockIdx.x * 128;

  v8f acc[2][4];
#pragma unroll
  for (int i = 0; i < 2; ++i)
#pragma unroll
    for (int j = 0; j < 4; ++j) acc[i][j] = {};

  const unsigned atOff0 = (unsigned)(uintptr_t)(void*)&At[0][0];
  const unsigned atOff1 = (unsigned)(uintptr_t)(void*)&At[1][0];
  const unsigned btOff0 = (unsigned)(uintptr_t)(void*)&Bt[0][0];
  const unsigned btOff1 = (unsigned)(uintptr_t)(void*)&Bt[1][0];

  // prologue: fetch K-tile 0 into buffer 0
  if (tid < 32) {
    tdm_load_2d_bf16(atOff0, A + (size_t)m0 * DD, DD, BT, 32, 128, DD);
    tdm_load_2d_bf16(btOff0, W + (size_t)n0 * DD, DD, DD, 32, 128, DD);
  }

  int buf = 0;
  for (int k0 = 0; k0 < DD; k0 += 32) {
    if (tid < 32) {
      if (k0 + 32 < DD) {
        // fetch next K-tile into the other buffer, then wait until the
        // two older TDM ops (current buffer) have retired.
        unsigned aoff = buf ? atOff0 : atOff1;
        unsigned boff = buf ? btOff0 : btOff1;
        tdm_load_2d_bf16(aoff, A + (size_t)m0 * DD + k0 + 32, DD, BT, 32, 128, DD);
        tdm_load_2d_bf16(boff, W + (size_t)n0 * DD + k0 + 32, DD, DD, 32, 128, DD);
        __builtin_amdgcn_s_wait_tensorcnt(2);
      } else {
        __builtin_amdgcn_s_wait_tensorcnt(0);
      }
    }
    __syncthreads();

    const bf16* Ab = At[buf];
    const bf16* Bb = Bt[buf];
    v16bf af[2];
#pragma unroll
    for (int mt = 0; mt < 2; ++mt)
      af[mt] = *(const v16bf*)(Ab + (wm * 32 + mt * 16 + (lane & 15)) * 32 +
                               (lane >> 4) * 16);
#pragma unroll
    for (int nt = 0; nt < 4; ++nt) {
      v16bf bfv = *(const v16bf*)(Bb + (wn * 64 + nt * 16 + (lane & 15)) * 32 +
                                  (lane >> 4) * 16);
      acc[0][nt] = wmma_bf16(af[0], bfv, acc[0][nt]);
      acc[1][nt] = wmma_bf16(af[1], bfv, acc[1][nt]);
    }
    __syncthreads();
    buf ^= 1;
  }

  // epilogue: C/D layout — VGPR r holds (m = r + 8*(lane>>4), n = lane&15)
#pragma unroll
  for (int mt = 0; mt < 2; ++mt) {
#pragma unroll
    for (int nt = 0; nt < 4; ++nt) {
#pragma unroll
      for (int r = 0; r < 8; ++r) {
        int m = m0 + wm * 32 + mt * 16 + r + 8 * (lane >> 4);
        int n = n0 + wn * 64 + nt * 16 + (lane & 15);
        float v = (acc[mt][nt][r] + bias[n]) * scale;
        if (mode == 3) {
          out_f[(size_t)m * DD + n] = v;
        } else if (mode == 2) {
          int b = m / TT, t = m % TT;
          int h = n >> 6, d = n & 63;
          out_bf[(((size_t)b * HH + h) * HD + d) * TT + t] = (bf16)v;
        } else {
          out_bf[(size_t)m * DD + n] = (bf16)v;
        }
      }
    }
  }
}

// ---------------------------------------------------------------- attention
// one workgroup = (b, h, 32 query rows). LDS: 32x2048 fp32 scores + 32x64 bf16 q.
__global__ __launch_bounds__(256) void attn_kernel(
    const bf16* __restrict__ qp, const bf16* __restrict__ kp,
    const bf16* __restrict__ vt, bf16* __restrict__ ob,
    float* __restrict__ wout) {
  extern __shared__ char smem[];
  float* S = (float*)smem;                               // 32*2048 fp32
  bf16* qt = (bf16*)(smem + 32 * 2048 * sizeof(float));  // 32*64 bf16

  const int tid = threadIdx.x;
  const int lane = tid & 31;
  const int wave = tid >> 5;
  const int q0 = blockIdx.x * 32;
  const int h = blockIdx.y;
  const int b = blockIdx.z;

  // stage q tile (32 rows x 64 cols bf16) via one TDM descriptor
  if (tid < 32) {
    const bf16* src = qp + (size_t)(b * TT + q0) * DD + h * HD;
    tdm_load_2d_bf16((unsigned)(uintptr_t)(void*)qt, src, DD, BT, HD, 32, DD);
    __builtin_amdgcn_s_wait_tensorcnt(0);
  }
  __syncthreads();

  // q fragments held in registers for the whole score sweep
  v16bf aq[2][2];
#pragma unroll
  for (int mt = 0; mt < 2; ++mt)
#pragma unroll
    for (int kk = 0; kk < 2; ++kk)
      aq[mt][kk] = *(const v16bf*)(qt + (mt * 16 + (lane & 15)) * 64 + kk * 32 +
                                   (lane >> 4) * 16);

  // S[0:32, wave*256 : wave*256+256] = q_tile @ k_h^T
  const bf16* kbase = kp + (size_t)b * TT * DD + h * HD;
  for (int j = 0; j < 16; ++j) {
    int sb = wave * 256 + j * 16;
    v8f a0 = {}, a1 = {};
#pragma unroll
    for (int kk = 0; kk < 2; ++kk) {
      v16bf bv = *(const v16bf*)(kbase + (size_t)(sb + (lane & 15)) * DD +
                                 kk * 32 + (lane >> 4) * 16);
      a0 = wmma_bf16(aq[0][kk], bv, a0);
      a1 = wmma_bf16(aq[1][kk], bv, a1);
    }
    int nn = sb + (lane & 15);
#pragma unroll
    for (int r = 0; r < 8; ++r) {
      S[(r + 8 * (lane >> 4)) * 2048 + nn] = a0[r];
      S[(16 + r + 8 * (lane >> 4)) * 2048 + nn] = a1[r];
    }
  }
  __syncthreads();

  // softmax: 4 rows per wave, strided lanes + shfl_xor wave32 reduction
  for (int rr = 0; rr < 4; ++rr) {
    int row = wave * 4 + rr;
    float* Sr = S + row * 2048;
    float mx = -3.0e38f;
    for (int c = lane; c < 2048; c += 32) mx = fmaxf(mx, Sr[c]);
#pragma unroll
    for (int off = 16; off > 0; off >>= 1) mx = fmaxf(mx, __shfl_xor(mx, off, 32));
    float sum = 0.0f;
    for (int c = lane; c < 2048; c += 32) {
      float e = __expf(Sr[c] - mx);
      Sr[c] = e;
      sum += e;
    }
#pragma unroll
    for (int off = 16; off > 0; off >>= 1) sum += __shfl_xor(sum, off, 32);
    float inv = 1.0f / sum;
    float* wp = wout + ((size_t)b * TT + q0 + row) * TT;
    for (int c = lane; c < 2048; c += 32) {
      float a = Sr[c] * inv;
      Sr[c] = a;
      atomicAdd(wp + c, a * 0.0625f);  // mean over 16 heads
    }
  }
  __syncthreads();

  // O_tile = A(32x2048) @ V_h(2048x64); one 16x16 tile per wave, K = 2048
  const bf16* vbase = vt + ((size_t)(b * HH + h) * HD) * TT;
  const int mt = wave >> 2, nt = wave & 3;
  v8f acc = {};
  for (int ss = 0; ss < 64; ++ss) {
    v16f sf = *(const v16f*)(S + (mt * 16 + (lane & 15)) * 2048 + ss * 32 +
                             (lane >> 4) * 16);
    v16bf av;
#pragma unroll
    for (int j = 0; j < 16; ++j) av[j] = (bf16)sf[j];
    v16bf bv = *(const v16bf*)(vbase + (size_t)(nt * 16 + (lane & 15)) * TT +
                               ss * 32 + (lane >> 4) * 16);
    acc = wmma_bf16(av, bv, acc);
  }
#pragma unroll
  for (int r = 0; r < 8; ++r) {
    int m = q0 + mt * 16 + r + 8 * (lane >> 4);
    int d = h * HD + nt * 16 + (lane & 15);
    ob[(size_t)(b * TT + m) * DD + d] = (bf16)acc[r];
  }
}

// ---------------------------------------------------------------- launch
extern "C" void kernel_launch(void* const* d_in, const int* in_sizes, int n_in,
                              void* d_out, int out_size, void* d_ws,
                              size_t ws_size, hipStream_t stream) {
  const float* Q  = (const float*)d_in[0];
  const float* K  = (const float*)d_in[1];
  const float* V  = (const float*)d_in[2];
  const float* Wq = (const float*)d_in[3];
  const float* bq = (const float*)d_in[4];
  const float* Wk = (const float*)d_in[5];
  const float* bk = (const float*)d_in[6];
  const float* Wv = (const float*)d_in[7];
  const float* bv = (const float*)d_in[8];
  const float* Wo = (const float*)d_in[9];
  const float* bo = (const float*)d_in[10];

  float* outO = (float*)d_out;                 // [B,T,D]
  float* outW = outO + (size_t)BT * DD;        // [B,T,T]

  char* ws = (char*)d_ws;
  size_t off = 0;
  auto carve = [&](size_t bytes) -> char* {
    char* p = ws + off;
    off += (bytes + 255) & ~(size_t)255;
    return p;
  };
  bf16* Qb  = (bf16*)carve((size_t)BT * DD * 2);
  bf16* Kb  = (bf16*)carve((size_t)BT * DD * 2);
  bf16* Vb  = (bf16*)carve((size_t)BT * DD * 2);
  bf16* Wqb = (bf16*)carve((size_t)DD * DD * 2);
  bf16* Wkb = (bf16*)carve((size_t)DD * DD * 2);
  bf16* Wvb = (bf16*)carve((size_t)DD * DD * 2);
  bf16* Wob = (bf16*)carve((size_t)DD * DD * 2);
  bf16* qpj = (bf16*)carve((size_t)BT * DD * 2);  // projected q (pre-scaled)
  bf16* kpj = (bf16*)carve((size_t)BT * DD * 2);  // projected k
  bf16* vtb = (bf16*)carve((size_t)BT * DD * 2);  // projected v, [b,h,d,t]
  bf16* obf = (bf16*)carve((size_t)BT * DD * 2);  // attention output, bf16

  cvt_bf16_kernel<<<4096, 256, 0, stream>>>(Q, Qb, BT * DD);
  cvt_bf16_kernel<<<4096, 256, 0, stream>>>(K, Kb, BT * DD);
  cvt_bf16_kernel<<<4096, 256, 0, stream>>>(V, Vb, BT * DD);
  cvt_bf16_kernel<<<1024, 256, 0, stream>>>(Wq, Wqb, DD * DD);
  cvt_bf16_kernel<<<1024, 256, 0, stream>>>(Wk, Wkb, DD * DD);
  cvt_bf16_kernel<<<1024, 256, 0, stream>>>(Wv, Wvb, DD * DD);
  cvt_bf16_kernel<<<1024, 256, 0, stream>>>(Wo, Wob, DD * DD);

  dim3 gg(DD / 128, BT / 128);  // (8, 64)
  gemm_proj_kernel<<<gg, 256, 0, stream>>>(Qb, Wqb, bq, qpj, nullptr, 0.125f, 0);
  gemm_proj_kernel<<<gg, 256, 0, stream>>>(Kb, Wkb, bk, kpj, nullptr, 1.0f, 0);
  gemm_proj_kernel<<<gg, 256, 0, stream>>>(Vb, Wvb, bv, vtb, nullptr, 1.0f, 2);

  zero_kernel<<<4096, 256, 0, stream>>>(outW, BT * TT);

  dim3 ag(TT / 32, HH, BB);
  size_t smem = (size_t)32 * 2048 * 4 + (size_t)32 * 64 * 2;  // 260 KB
  attn_kernel<<<ag, 256, smem, stream>>>(qpj, kpj, vtb, obf, outW);

  gemm_proj_kernel<<<gg, 256, 0, stream>>>(obf, Wob, bo, nullptr, outO, 1.0f, 3);
}
